// SelfAttn2_12524124635343
// MI455X (gfx1250) — matrix operand order repaired and checked by reading, not compiled, not run on hardware
//
#include <hip/hip_runtime.h>

typedef __bf16 bf16;
typedef __attribute__((ext_vector_type(16))) __bf16 v16bf;
typedef __attribute__((ext_vector_type(8)))  float   v8f;

#define LDSS 40   // LDS row stride in halves (80B -> 16B aligned rows)
#define BM 128
#define BN 128
#define BK 32
#define BUFH ((BM + BN) * LDSS)        // halves per double-buffer stage

static __device__ inline bf16 to_bf16(float f) {
    unsigned u = __builtin_bit_cast(unsigned, f);
    unsigned r = (u + 0x7FFFu + ((u >> 16) & 1u)) >> 16;
    unsigned short s = (unsigned short)r;
    return __builtin_bit_cast(bf16, s);
}

union Frag { v16bf v; unsigned int u[8]; };
union U4h  { uint4 q; bf16 h[8]; };

// ---------------------------------------------------------------------------
// Generic bf16 GEMM: C[z] = alpha * A[z%aMod] * B[z] (+bias[row]) (+gamma*.+resid)
// A: M x K row-major (bf16, lda).  B: if bT==0 K x N row-major (ldb=N stride),
// if bT==1 it is stored N x K row-major (ldb=K stride), i.e. B-transposed.
// Tiles staged with GLOBAL_LOAD_ASYNC_TO_LDS_B128 (ASYNCcnt), 2-deep LDS
// double buffer so global latency overlaps the WMMA phase; out-of-range
// rows/cols are clamped (outputs masked in the epilogue) -> branch-free.
// ---------------------------------------------------------------------------
__global__ __launch_bounds__(256)
void gemm_bf16_wmma(const bf16* __restrict__ A, const bf16* __restrict__ Bp,
                    float* __restrict__ outF, bf16* __restrict__ outB,
                    const float* __restrict__ bias,
                    const float* __restrict__ resid, const float* __restrict__ gammaPtr,
                    int M, int N, int K, int lda, int ldb,
                    long strideA, int aMod, long strideB, long strideC,
                    float alpha, int bT)
{
    __shared__ __align__(16) bf16 ldsbuf[2 * BUFH];   // 40 KB

    const int z = blockIdx.z;
    const bf16* Az = A + (long)(z % aMod) * strideA;
    const bf16* Bz = Bp + (long)z * strideB;
    const long  cbase = (long)z * strideC;

    const int tileM = blockIdx.y * BM;
    const int tileN = blockIdx.x * BN;

    const int tid  = threadIdx.x;
    const int lane = tid & 31;
    const int wid  = tid >> 5;
    const int waveM = (wid >> 1) * 32;  // 4 waves along M
    const int waveN = (wid & 1) * 64;   // 2 waves along N
    const int l15 = lane & 15;
    const int lh  = lane >> 4;          // 0: lanes 0-15, 1: lanes 16-31

    // chunk mapping for 128 x 32-half tiles: 512 chunks of 8 halves, 2/thread
    const int c0row = tid >> 2, ccol = (tid & 3) << 3;
    const int c1row = (tid + 256) >> 2;

    // clamped source rows (k-invariant)
    int gmA0 = tileM + c0row; if (gmA0 > M - 1) gmA0 = M - 1;
    int gmA1 = tileM + c1row; if (gmA1 > M - 1) gmA1 = M - 1;
    int gnB0 = tileN + c0row; if (gnB0 > N - 1) gnB0 = N - 1;
    int gnB1 = tileN + c1row; if (gnB1 > N - 1) gnB1 = N - 1;
    const int scKrow = tid >> 4, scNc = (tid & 15) << 3;           // scatter path
    int scGn = tileN + scNc; if (scGn > N - 8) scGn = N - 8;

    // stage k-tile (k0) into double-buffer slot s
    auto stage = [&](int k0, int s) {
        const unsigned baseA = (unsigned)(s * BUFH * 2);
        const unsigned baseB = baseA + (unsigned)(BM * LDSS * 2);
        {
            unsigned long long ga0 =
                (unsigned long long)(uintptr_t)(Az + (long)gmA0 * lda + k0 + ccol);
            unsigned long long ga1 =
                (unsigned long long)(uintptr_t)(Az + (long)gmA1 * lda + k0 + ccol);
            unsigned lo0 = baseA + (unsigned)((c0row * LDSS + ccol) * 2);
            unsigned lo1 = baseA + (unsigned)((c1row * LDSS + ccol) * 2);
            asm volatile("global_load_async_to_lds_b128 %0, %1, off"
                         :: "v"(lo0), "v"(ga0) : "memory");
            asm volatile("global_load_async_to_lds_b128 %0, %1, off"
                         :: "v"(lo1), "v"(ga1) : "memory");
        }
        if (bT) {
            unsigned long long ga0 =
                (unsigned long long)(uintptr_t)(Bz + (long)gnB0 * ldb + k0 + ccol);
            unsigned long long ga1 =
                (unsigned long long)(uintptr_t)(Bz + (long)gnB1 * ldb + k0 + ccol);
            unsigned lo0 = baseB + (unsigned)((c0row * LDSS + ccol) * 2);
            unsigned lo1 = baseB + (unsigned)((c1row * LDSS + ccol) * 2);
            asm volatile("global_load_async_to_lds_b128 %0, %1, off"
                         :: "v"(lo0), "v"(ga0) : "memory");
            asm volatile("global_load_async_to_lds_b128 %0, %1, off"
                         :: "v"(lo1), "v"(ga1) : "memory");
        } else {
            // source is K x N: load contiguous along N, scatter-transpose to LDS
            bf16* sBp = ldsbuf + s * BUFH + BM * LDSS;
#pragma unroll
            for (int it = 0; it < 2; ++it) {
                int krow = scKrow + it * 16;
                U4h u;
                u.q = *reinterpret_cast<const uint4*>(Bz + (long)(k0 + krow) * ldb + scGn);
#pragma unroll
                for (int j = 0; j < 8; ++j)
                    sBp[(scNc + j) * LDSS + krow] = u.h[j];
            }
        }
    };

    v8f acc[2][4];
#pragma unroll
    for (int i = 0; i < 2; ++i)
#pragma unroll
        for (int j = 0; j < 4; ++j)
#pragma unroll
            for (int e = 0; e < 8; ++e) acc[i][j][e] = 0.0f;

    const int nsteps = K / BK;     // all call sites have K % 32 == 0
    stage(0, 0);
    asm volatile("s_wait_asynccnt 0x0" ::: "memory");
    __syncthreads();

    for (int i = 0; i < nsteps; ++i) {
        const int cur = i & 1;
        if (i + 1 < nsteps) stage((i + 1) * BK, cur ^ 1);

        const bf16* sA = ldsbuf + cur * BUFH;
        const bf16* sB = sA + BM * LDSS;

        Frag af[2], bfv[4];
#pragma unroll
        for (int mt = 0; mt < 2; ++mt) {
            const unsigned int* ap = reinterpret_cast<const unsigned int*>(
                &sA[(waveM + mt * 16 + l15) * LDSS + lh * 8]);
            af[mt].u[0] = ap[0];  af[mt].u[1] = ap[1];
            af[mt].u[2] = ap[2];  af[mt].u[3] = ap[3];
            af[mt].u[4] = ap[8];  af[mt].u[5] = ap[9];
            af[mt].u[6] = ap[10]; af[mt].u[7] = ap[11];
        }
#pragma unroll
        for (int nt = 0; nt < 4; ++nt) {
            const unsigned int* bp = reinterpret_cast<const unsigned int*>(
                &sB[(waveN + nt * 16 + l15) * LDSS + lh * 8]);
#pragma unroll
            for (int j = 0; j < 8; ++j) bfv[nt].u[j] = bp[j < 4 ? j : j + 4];
        }
#pragma unroll
        for (int mt = 0; mt < 2; ++mt)
#pragma unroll
            for (int nt = 0; nt < 4; ++nt)
                acc[mt][nt] = __builtin_amdgcn_wmma_f32_16x16x32_bf16(
                    false, af[mt].v, false, bfv[nt].v, (short)0, acc[mt][nt],
                    false, false);

        if (i + 1 < nsteps)
            asm volatile("s_wait_asynccnt 0x0" ::: "memory");
        __syncthreads();
    }

    const float gmv = gammaPtr ? gammaPtr[0] : 0.0f;
#pragma unroll
    for (int mt = 0; mt < 2; ++mt)
#pragma unroll
        for (int nt = 0; nt < 4; ++nt)
#pragma unroll
            for (int i = 0; i < 8; ++i) {
                int row = tileM + waveM + mt * 16 + i + lh * 8;
                int col = tileN + waveN + nt * 16 + l15;
                if (row < M && col < N) {
                    float v = acc[mt][nt][i] * alpha;
                    if (bias)  v += bias[row];
                    long idx = cbase + (long)row * N + col;
                    if (resid) v = gmv * v + resid[idx];
                    if (outF)  outF[idx] = v;
                    if (outB)  outB[idx] = to_bf16(v);
                }
            }
}

// ---------------------------------------------------------------------------
// f32 -> bf16 conversion (vec4)
// ---------------------------------------------------------------------------
__global__ __launch_bounds__(256)
void cvt4_k(const float* __restrict__ s, bf16* __restrict__ d, long n4)
{
    long i = (long)blockIdx.x * 256 + threadIdx.x;
    if (i >= n4) return;
    float4 v = reinterpret_cast<const float4*>(s)[i];
    union { bf16 h[4]; unsigned long long u; } p;
    p.h[0] = to_bf16(v.x); p.h[1] = to_bf16(v.y);
    p.h[2] = to_bf16(v.z); p.h[3] = to_bf16(v.w);
    reinterpret_cast<unsigned long long*>(d)[i] = p.u;
}

// ---------------------------------------------------------------------------
// L2-normalize Q per (b,g,hw) over the 64 per-group channels -> bf16
// ---------------------------------------------------------------------------
__global__ __launch_bounds__(256)
void qnorm_k(const float* __restrict__ Q, bf16* __restrict__ Qn, int hw)
{
    int h = blockIdx.x * 256 + threadIdx.x;
    if (h >= hw) return;
    long base = (long)blockIdx.y * 64 * hw + h;
    float ss = 0.0f;
    for (int c = 0; c < 64; ++c) { float v = Q[base + (long)c * hw]; ss += v * v; }
    float inv = 1.0f / fmaxf(sqrtf(ss), 1e-12f);
    for (int c = 0; c < 64; ++c)
        Qn[base + (long)c * hw] = to_bf16(Q[base + (long)c * hw] * inv);
}

// ---------------------------------------------------------------------------
// Normalize centroid weights (128 rows x 64) -> f32 + bf16
// ---------------------------------------------------------------------------
__global__ __launch_bounds__(128)
void wgnorm_k(const float* __restrict__ w, float* __restrict__ Wf, bf16* __restrict__ Wb)
{
    int r = threadIdx.x;
    float ss = 0.0f;
    for (int i = 0; i < 64; ++i) { float v = w[r * 64 + i]; ss += v * v; }
    float inv = 1.0f / fmaxf(sqrtf(ss), 1e-12f);
    for (int i = 0; i < 64; ++i) {
        float v = w[r * 64 + i] * inv;
        Wf[r * 64 + i] = v;
        Wb[r * 64 + i] = to_bf16(v);
    }
}

// ---------------------------------------------------------------------------
// Bm = softmax(Wg Wg^T * 14) then Binv = inv(Bm): Gauss-Jordan in LDS, 1 blk/group
// ---------------------------------------------------------------------------
__global__ __launch_bounds__(64)
void binv_k(const float* __restrict__ Wf, float* __restrict__ Binv)
{
    __shared__ float aug[64][129];
    __shared__ float pivinv;
    int g = blockIdx.x, t = threadIdx.x;
    const float* Wg = Wf + g * 64 * 64;
    float rowv[64];
    float mx = -3.4e38f;
    for (int k = 0; k < 64; ++k) {
        float d = 0.0f;
        for (int c = 0; c < 64; ++c) d += Wg[t * 64 + c] * Wg[k * 64 + c];
        d *= 14.0f;
        rowv[k] = d; mx = fmaxf(mx, d);
    }
    float s = 0.0f;
    for (int k = 0; k < 64; ++k) { rowv[k] = __expf(rowv[k] - mx); s += rowv[k]; }
    float inv = 1.0f / s;
    for (int k = 0; k < 64; ++k) aug[t][k] = rowv[k] * inv;
    for (int k = 0; k < 64; ++k) aug[t][64 + k] = (k == t) ? 1.0f : 0.0f;
    __syncthreads();
    for (int k = 0; k < 64; ++k) {
        if (t == 0) pivinv = 1.0f / aug[k][k];
        __syncthreads();
        aug[k][t] *= pivinv;
        aug[k][t + 64] *= pivinv;
        __syncthreads();
        if (t != k) {
            float f = aug[t][k];
            for (int c = 0; c < 128; ++c) aug[t][c] -= f * aug[k][c];
        }
        __syncthreads();
    }
    for (int k = 0; k < 64; ++k) Binv[((long)g * 64 + t) * 64 + k] = aug[t][64 + k];
}

// ---------------------------------------------------------------------------
// softmax over centroids (axis m, 64 entries, stride hw) per (z,h) column
// ---------------------------------------------------------------------------
__global__ __launch_bounds__(256)
void softmax_cols_k(const float* __restrict__ src, bf16* __restrict__ dst, int hw)
{
    int h = blockIdx.x * 256 + threadIdx.x;
    if (h >= hw) return;
    long base = (long)blockIdx.y * 64 * hw + h;
    float mx = -3.4e38f;
    for (int m = 0; m < 64; ++m) mx = fmaxf(mx, src[base + (long)m * hw]);
    float s = 0.0f;
    for (int m = 0; m < 64; ++m) s += __expf(src[base + (long)m * hw] - mx);
    float inv = 1.0f / s;
    for (int m = 0; m < 64; ++m)
        dst[base + (long)m * hw] = to_bf16(__expf(src[base + (long)m * hw] - mx) * inv);
}

// ---------------------------------------------------------------------------
// softmax over spatial (one block per (z,m) row of ncols)
// ---------------------------------------------------------------------------
__global__ __launch_bounds__(256)
void softmax_rows_k(const float* __restrict__ src, bf16* __restrict__ dst, int ncols)
{
    __shared__ float red[256];
    long base = (long)blockIdx.x * ncols;
    int tid = threadIdx.x;
    float mx = -3.4e38f;
    for (int c = tid; c < ncols; c += 256) mx = fmaxf(mx, src[base + c]);
    red[tid] = mx; __syncthreads();
    for (int s = 128; s > 0; s >>= 1) { if (tid < s) red[tid] = fmaxf(red[tid], red[tid + s]); __syncthreads(); }
    mx = red[0]; __syncthreads();
    float sum = 0.0f;
    for (int c = tid; c < ncols; c += 256) sum += __expf(src[base + c] - mx);
    red[tid] = sum; __syncthreads();
    for (int s = 128; s > 0; s >>= 1) { if (tid < s) red[tid] += red[tid + s]; __syncthreads(); }
    float inv = 1.0f / red[0];
    for (int c = tid; c < ncols; c += 256)
        dst[base + c] = to_bf16(__expf(src[base + c] - mx) * inv);
}

// ---------------------------------------------------------------------------
// O2[z][c][k] = sum_m O[z][c][m] * Binv[g][k][m]  (g = z & 1), -> bf16
// ---------------------------------------------------------------------------
__global__ __launch_bounds__(256)
void obinv_k(const float* __restrict__ O, const float* __restrict__ Binv, bf16* __restrict__ O2)
{
    long idx = (long)blockIdx.x * 256 + threadIdx.x;   // 16*256*64 total
    int k = (int)(idx & 63);
    long r = idx >> 6;
    int c = (int)(r & 255);
    long z = r >> 8;
    int g = (int)(z & 1);
    const float* Or = O + (z * 256 + c) * 64;
    const float* Br = Binv + ((long)g * 64 + k) * 64;
    float s = 0.0f;
    for (int m = 0; m < 64; ++m) s += Or[m] * Br[m];
    O2[idx] = to_bf16(s);
}

__global__ void tail_k(float* out, long idx) { out[idx] = 0.0f; }

// ---------------------------------------------------------------------------
extern "C" void kernel_launch(void* const* d_in, const int* in_sizes, int n_in,
                              void* d_out, int out_size, void* d_ws, size_t ws_size,
                              hipStream_t stream)
{
    (void)in_sizes; (void)n_in; (void)out_size; (void)ws_size;
    constexpr int HW = 96 * 96;         // 9216
    constexpr int Cc = 512, C4 = 128;
    constexpr long NX = 8L * Cc * HW;   // 37,748,736  x / V / Of elements
    constexpr long NQ = 8L * C4 * HW;   //  9,437,184  Q / Ml / Cm / A elements

    const float* x     = (const float*)d_in[0];
    const float* qw    = (const float*)d_in[1];
    const float* qb    = (const float*)d_in[2];
    const float* vw    = (const float*)d_in[3];
    const float* vb    = (const float*)d_in[4];
    const float* ow    = (const float*)d_in[5];
    const float* ob    = (const float*)d_in[6];
    const float* w     = (const float*)d_in[7];
    const float* gamma = (const float*)d_in[8];
    float* out = (float*)d_out;

    char* base = (char*)d_ws;
    size_t off = 0;
    auto alloc = [&](size_t bytes) { size_t o = off; off = (off + bytes + 255) & ~(size_t)255; return o; };
    bf16*  xb   = (bf16*)(base + alloc(NX * 2));         // x bf16; later reused as Of bf16
    bf16*  Vb   = (bf16*)(base + alloc(NX * 2));
    float* Qf   = (float*)(base + alloc(NQ * 4));        // Q f32; later reused as Ml f32
    bf16*  Qnb  = (bf16*)(base + alloc(NQ * 2));
    bf16*  Cmb  = (bf16*)(base + alloc(NQ * 2));
    bf16*  Ab   = (bf16*)(base + alloc(NQ * 2));
    bf16*  qwb  = (bf16*)(base + alloc(128L * 512 * 2));
    bf16*  vwb  = (bf16*)(base + alloc(512L * 512 * 2));
    bf16*  owb  = (bf16*)(base + alloc(512L * 512 * 2));
    float* Wgf  = (float*)(base + alloc(128L * 64 * 4));
    bf16*  Wgb  = (bf16*)(base + alloc(128L * 64 * 2));
    float* Binv = (float*)(base + alloc(2L * 64 * 64 * 4));
    float* Obuf = (float*)(base + alloc(16L * 256 * 64 * 4));
    bf16*  O2b  = (bf16*)(base + alloc(16L * 256 * 64 * 2));
    float* Ml   = Qf;    // Q dead after qnorm
    bf16*  Ofb  = xb;    // x bf16 dead after V gemm

    const int BIG = 1 << 30;

    // 1) conversions + centroid prep
    cvt4_k<<<(unsigned)(NX / 4 / 256), 256, 0, stream>>>(x, xb, NX / 4);
    cvt4_k<<<(128 * 512 / 4) / 256, 256, 0, stream>>>(qw, qwb, 128L * 512 / 4);
    cvt4_k<<<(512 * 512 / 4) / 256, 256, 0, stream>>>(vw, vwb, 512L * 512 / 4);
    cvt4_k<<<(512 * 512 / 4) / 256, 256, 0, stream>>>(ow, owb, 512L * 512 / 4);
    wgnorm_k<<<1, 128, 0, stream>>>(w, Wgf, Wgb);
    binv_k<<<2, 64, 0, stream>>>(Wgf, Binv);

    // 2) Q = qw * x + qb  (f32 out)
    gemm_bf16_wmma<<<dim3(HW / 128, 1, 8), 256, 0, stream>>>(
        qwb, xb, Qf, nullptr, qb, nullptr, nullptr,
        128, HW, 512, 512, HW, 0, 1, (long)Cc * HW, (long)C4 * HW, 1.0f, 0);

    // 3) group-L2-normalize Q -> bf16
    qnorm_k<<<dim3(HW / 256, 16), 256, 0, stream>>>(Qf, Qnb, HW);

    // 4) V = vw * x + vb  (bf16 out)
    gemm_bf16_wmma<<<dim3(HW / 128, 4, 8), 256, 0, stream>>>(
        vwb, xb, nullptr, Vb, vb, nullptr, nullptr,
        512, HW, 512, 512, HW, 0, 1, (long)Cc * HW, (long)Cc * HW, 1.0f, 0);

    // 5) Ml = 14 * Wgn * Qn  per (b,g)   (f32 out, overwrites Q)
    gemm_bf16_wmma<<<dim3(HW / 128, 1, 16), 256, 0, stream>>>(
        Wgb, Qnb, Ml, nullptr, nullptr, nullptr, nullptr,
        64, HW, 64, 64, HW, 64L * 64, 2, 64L * HW, 64L * HW, 14.0f, 0);

    // 6) Cm = softmax over centroids;  7) A = softmax over spatial
    softmax_cols_k<<<dim3(HW / 256, 16), 256, 0, stream>>>(Ml, Cmb, HW);
    softmax_rows_k<<<16 * 64, 256, 0, stream>>>(Ml, Ab, HW);

    // 8) O = Vg * A^T  per (b,g):  M=256, N=64, K=HW (B transposed mode)
    gemm_bf16_wmma<<<dim3(1, 2, 16), 256, 0, stream>>>(
        Vb, Ab, Obuf, nullptr, nullptr, nullptr, nullptr,
        256, 64, HW, HW, HW, 256L * HW, BIG, 64L * HW, 256L * 64, 1.0f, 1);

    // 9) O2 = O * Binv^T  (bf16 out)
    obinv_k<<<(16 * 256 * 64) / 256, 256, 0, stream>>>(Obuf, Binv, O2b);

    // 10) Of = O2 * Cm  per (b,g)  (bf16 out, reuses xb)
    gemm_bf16_wmma<<<dim3(HW / 128, 2, 16), 256, 0, stream>>>(
        O2b, Cmb, nullptr, Ofb, nullptr, nullptr, nullptr,
        256, HW, 64, 64, HW, 256L * 64, BIG, 64L * HW, 256L * HW, 1.0f, 0);

    // 11) out = gamma * (ow * Of + ob) + x
    gemm_bf16_wmma<<<dim3(HW / 128, 4, 8), 256, 0, stream>>>(
        owb, Ofb, out, nullptr, ob, x, gamma,
        512, HW, 512, 512, HW, 0, 1, (long)Cc * HW, (long)Cc * HW, 1.0f, 0);

    // 12) trailing scalar output = 0
    tail_k<<<1, 1, 0, stream>>>(out, NX);
}
